// EGCL_16217796509989
// MI455X (gfx1250) — matrix operand order repaired and checked by reading, not compiled, not run on hardware
//
#include <hip/hip_runtime.h>
#include <math.h>

typedef _Float16 f16h;
typedef __attribute__((ext_vector_type(4)))  _Float16 v4h;
typedef __attribute__((ext_vector_type(8)))  _Float16 v8h;
typedef __attribute__((ext_vector_type(16))) _Float16 v16h;
typedef __attribute__((ext_vector_type(8)))  float    v8f;

#define NN 50000
#define NE 800000

__device__ __forceinline__ float silu_f(float x) {
    return x / (1.0f + __expf(-x));
}

// A fragment (16x32 f16) from LDS row-major tile.
// ISA 7.12.2: lane m = lane&15, half = lane>>4.
// element e=0..7  -> K = chunk*32 + half*8 + e          (one 16B run)
// element e=8..15 -> K = chunk*32 + 16 + half*8 + (e-8) (one 16B run)
__device__ __forceinline__ v16h load_a_lds(const f16h* tile, int stride, int mBase,
                                           int lane, int kChunk) {
    const int m = lane & 15;
    const int half = lane >> 4;
    const f16h* rowp = tile + (mBase + m) * stride + kChunk * 32 + half * 8;
    v8h lo = *(const v8h*)(rowp);
    v8h hi = *(const v8h*)(rowp + 16);
    return __builtin_shufflevector(lo, hi, 0, 1, 2, 3, 4, 5, 6, 7,
                                   8, 9, 10, 11, 12, 13, 14, 15);
}

// B fragment from pre-swizzled weights: one contiguous 32B load per lane.
__device__ __forceinline__ v16h load_b_packed(const f16h* __restrict__ Wp, int nkc,
                                              int ct, int kc, int lane) {
    const v16h* p = (const v16h*)Wp;
    return p[((ct * nkc + kc) << 5) + lane];
}

// Pack f32 weights W[K][ncols] into B-fragment order:
// out[((ct*nkc+kc)*32 + lane)*16 + e] = W[kc*32 + (e>>3)*16 + (lane>>4)*8 + (e&7)][ct*16 + (lane&15)]
__global__ void pack_b_f16(const float* __restrict__ W, f16h* __restrict__ out,
                           int K, int ncols) {
    int idx = blockIdx.x * blockDim.x + threadIdx.x;
    if (idx >= K * ncols) return;
    int e    = idx & 15;
    int lane = (idx >> 4) & 31;
    int rest = idx >> 9;            // ct*nkc + kc
    int nkc  = K >> 5;
    int kc   = rest % nkc;
    int ct   = rest / nkc;
    int k = kc * 32 + (e >> 3) * 16 + (lane >> 4) * 8 + (e & 7);
    int n = ct * 16 + (lane & 15);
    out[idx] = (f16h)W[k * ncols + n];
}

// ---------------------------------------------------------------------------
// Edge kernel: 64 edges/block, 8 waves.
//   wave w: rowTile = w>>1 (16 edges), colGroup = w&1 (4 col-tiles => 64 cols)
// gather -> L1 WMMA (+dist rank-1 folded into C init) -> SiLU -> L2 WMMA
//        -> SiLU -> atomic scatter into agg[row][*]
// ---------------------------------------------------------------------------
__global__ __launch_bounds__(256) void egcl_edge_kernel(
    const float* __restrict__ h, const int* __restrict__ row, const int* __restrict__ col,
    const float* __restrict__ dist,
    const float* __restrict__ eW1,   // f32 [129][128]; row 128 = dist term
    const float* __restrict__ eb1, const float* __restrict__ eb2,
    const f16h* __restrict__ eW1p,   // packed [8 ct][4 kc][32 lane][16]
    const f16h* __restrict__ eW2p,   // packed [8 ct][4 kc][32 lane][16]
    float* __restrict__ agg)         // [NN][128]
{
    __shared__ __align__(16) f16h Ain[64 * 136];
    __shared__ __align__(16) f16h E1 [64 * 136];
    __shared__ float distS[64];
    __shared__ int   rowS[64];
    __shared__ int   colS[64];

    const int tid  = threadIdx.x;
    const int lane = tid & 31;
    const int wave = tid >> 5;
    const int eBase = blockIdx.x * 64;

    __builtin_prefetch(eW1p, 0, 1);
    __builtin_prefetch(eW2p, 0, 1);

    if (tid < 64) {
        int eg = min(eBase + tid, NE - 1);
        rowS[tid]  = row[eg];
        colS[tid]  = col[eg];
        distS[tid] = dist[eg];
    }
    __syncthreads();

    // Gather h[row]|h[col] -> f16 LDS tile; float4 loads, v4h (b64) stores.
    for (int i = tid; i < 64 * 32; i += 256) {
        int el = i >> 5;
        int k4 = (i & 31) * 4;
        int node = (k4 < 64) ? rowS[el] : colS[el];
        const float4 v = *(const float4*)(h + node * 64 + (k4 & 63));
        v4h o;
        o[0] = (f16h)v.x; o[1] = (f16h)v.y; o[2] = (f16h)v.z; o[3] = (f16h)v.w;
        *(v4h*)(Ain + el * 136 + k4) = o;
    }
    __syncthreads();

    const int rowTile  = wave >> 1;        // 0..3
    const int mBase    = rowTile * 16;
    const int colGroup = wave & 1;         // 0..1
    const int nLane    = lane & 15;
    const int mOff     = 8 * (lane >> 4);

    // ---- Layer 1: acc = eb1 + dist * eW1[128][:]  then += feats @ eW1[0:128]
    const float* w1last = eW1 + 128 * 128;
    v8f acc[4];
#pragma unroll
    for (int t = 0; t < 4; ++t) {
        int nBase = (colGroup * 4 + t) * 16;
        float bias = eb1[nBase + nLane];
        float wl   = w1last[nBase + nLane];
#pragma unroll
        for (int r = 0; r < 8; ++r)
            acc[t][r] = bias + distS[mBase + r + mOff] * wl;
    }
#pragma unroll
    for (int kc = 0; kc < 4; ++kc) {
        v16h a = load_a_lds(Ain, 136, mBase, lane, kc);
#pragma unroll
        for (int t = 0; t < 4; ++t) {
            v16h b = load_b_packed(eW1p, 4, colGroup * 4 + t, kc, lane);
            acc[t] = __builtin_amdgcn_wmma_f32_16x16x32_f16(
                false, a, false, b, (short)0, acc[t], false, false);
        }
    }
#pragma unroll
    for (int t = 0; t < 4; ++t) {
        int nBase = (colGroup * 4 + t) * 16;
#pragma unroll
        for (int r = 0; r < 8; ++r)
            E1[(mBase + r + mOff) * 136 + nBase + nLane] = (f16h)silu_f(acc[t][r]);
    }
    __syncthreads();

    // ---- Layer 2 + scatter
    v8f acc2[4];
#pragma unroll
    for (int t = 0; t < 4; ++t) {
        int nBase = (colGroup * 4 + t) * 16;
        float bias = eb2[nBase + nLane];
#pragma unroll
        for (int r = 0; r < 8; ++r) acc2[t][r] = bias;
    }
#pragma unroll
    for (int kc = 0; kc < 4; ++kc) {
        v16h a = load_a_lds(E1, 136, mBase, lane, kc);
#pragma unroll
        for (int t = 0; t < 4; ++t) {
            v16h b = load_b_packed(eW2p, 4, colGroup * 4 + t, kc, lane);
            acc2[t] = __builtin_amdgcn_wmma_f32_16x16x32_f16(
                false, a, false, b, (short)0, acc2[t], false, false);
        }
    }
#pragma unroll
    for (int t = 0; t < 4; ++t) {
        int nBase = (colGroup * 4 + t) * 16;
#pragma unroll
        for (int r = 0; r < 8; ++r) {
            int m = mBase + r + mOff;
            if (eBase + m < NE) {
                float val = silu_f(acc2[t][r]);
                atomicAdd(&agg[rowS[m] * 128 + nBase + nLane], val);
            }
        }
    }
}

// ---------------------------------------------------------------------------
// Node kernel: 64 nodes/block; node_in = [h | agg] (K=192, 6 WMMA chunks)
// ---------------------------------------------------------------------------
__global__ __launch_bounds__(256) void egcl_node_kernel(
    const float* __restrict__ h, const float* __restrict__ agg,
    const float* __restrict__ nb1, const float* __restrict__ nb2,
    const f16h* __restrict__ nW1p,   // packed [8 ct][6 kc][32][16]
    const f16h* __restrict__ nW2p,   // packed [4 ct][4 kc][32][16]
    float* __restrict__ out)         // [NN][64]
{
    __shared__ __align__(16) f16h Ain[64 * 208];  // K=192, stride 208 (416B = 16*26)
    __shared__ __align__(16) f16h T1 [64 * 136];

    const int tid  = threadIdx.x;
    const int lane = tid & 31;
    const int wave = tid >> 5;
    const int nodeBase = blockIdx.x * 64;

    __builtin_prefetch(nW1p, 0, 1);
    __builtin_prefetch(nW2p, 0, 1);

    for (int i = tid; i < 64 * 48; i += 256) {
        int el = i / 48;
        int k4 = (i - el * 48) * 4;
        int node = min(nodeBase + el, NN - 1);
        const float* src = (k4 < 64) ? (h + node * 64 + k4)
                                     : (agg + node * 128 + (k4 - 64));
        const float4 v = *(const float4*)src;
        v4h o;
        o[0] = (f16h)v.x; o[1] = (f16h)v.y; o[2] = (f16h)v.z; o[3] = (f16h)v.w;
        *(v4h*)(Ain + el * 208 + k4) = o;
    }
    __syncthreads();

    const int rowTile  = wave >> 1;
    const int mBase    = rowTile * 16;
    const int colGroup = wave & 1;
    const int nLane    = lane & 15;
    const int mOff     = 8 * (lane >> 4);

    // ---- Layer 1 (K=192)
    v8f acc[4];
#pragma unroll
    for (int t = 0; t < 4; ++t) {
        int nBase = (colGroup * 4 + t) * 16;
        float bias = nb1[nBase + nLane];
#pragma unroll
        for (int r = 0; r < 8; ++r) acc[t][r] = bias;
    }
#pragma unroll
    for (int kc = 0; kc < 6; ++kc) {
        v16h a = load_a_lds(Ain, 208, mBase, lane, kc);
#pragma unroll
        for (int t = 0; t < 4; ++t) {
            v16h b = load_b_packed(nW1p, 6, colGroup * 4 + t, kc, lane);
            acc[t] = __builtin_amdgcn_wmma_f32_16x16x32_f16(
                false, a, false, b, (short)0, acc[t], false, false);
        }
    }
#pragma unroll
    for (int t = 0; t < 4; ++t) {
        int nBase = (colGroup * 4 + t) * 16;
#pragma unroll
        for (int r = 0; r < 8; ++r)
            T1[(mBase + r + mOff) * 136 + nBase + nLane] = (f16h)silu_f(acc[t][r]);
    }
    __syncthreads();

    // ---- Layer 2 (K=128 -> 64 cols; 2 col-tiles/wave), linear
    v8f acc2[2];
#pragma unroll
    for (int t = 0; t < 2; ++t) {
        int nBase = (colGroup * 2 + t) * 16;
        float bias = nb2[nBase + nLane];
#pragma unroll
        for (int r = 0; r < 8; ++r) acc2[t][r] = bias;
    }
#pragma unroll
    for (int kc = 0; kc < 4; ++kc) {
        v16h a = load_a_lds(T1, 136, mBase, lane, kc);
#pragma unroll
        for (int t = 0; t < 2; ++t) {
            v16h b = load_b_packed(nW2p, 4, colGroup * 2 + t, kc, lane);
            acc2[t] = __builtin_amdgcn_wmma_f32_16x16x32_f16(
                false, a, false, b, (short)0, acc2[t], false, false);
        }
    }
#pragma unroll
    for (int t = 0; t < 2; ++t) {
        int nBase = (colGroup * 2 + t) * 16;
#pragma unroll
        for (int r = 0; r < 8; ++r) {
            int m = mBase + r + mOff;
            int node = nodeBase + m;
            if (node < NN) out[node * 64 + nBase + nLane] = acc2[t][r];
        }
    }
}

extern "C" void kernel_launch(void* const* d_in, const int* in_sizes, int n_in,
                              void* d_out, int out_size, void* d_ws, size_t ws_size,
                              hipStream_t stream) {
    const float* h    = (const float*)d_in[0];
    const int*   row  = (const int*)  d_in[1];
    const int*   col  = (const int*)  d_in[2];
    const float* dist = (const float*)d_in[3];
    const float* eW1  = (const float*)d_in[4];
    const float* eb1  = (const float*)d_in[5];
    const float* eW2  = (const float*)d_in[6];
    const float* eb2  = (const float*)d_in[7];
    const float* nW1  = (const float*)d_in[8];
    const float* nb1  = (const float*)d_in[9];
    const float* nW2  = (const float*)d_in[10];
    const float* nb2  = (const float*)d_in[11];

    // Workspace: agg f32 [NN][128], then packed f16 weights.
    float* agg  = (float*)d_ws;
    f16h*  eW1p = (f16h*)(agg + (size_t)NN * 128);
    f16h*  eW2p = eW1p + 128 * 128;
    f16h*  nW1p = eW2p + 128 * 128;
    f16h*  nW2p = nW1p + 192 * 128;

    hipMemsetAsync(agg, 0, (size_t)NN * 128 * sizeof(float), stream);

    pack_b_f16<<<(128 * 128 + 255) / 256, 256, 0, stream>>>(eW1, eW1p, 128, 128);
    pack_b_f16<<<(128 * 128 + 255) / 256, 256, 0, stream>>>(eW2, eW2p, 128, 128);
    pack_b_f16<<<(192 * 128 + 255) / 256, 256, 0, stream>>>(nW1, nW1p, 192, 128);
    pack_b_f16<<<(128 * 64  + 255) / 256, 256, 0, stream>>>(nW2, nW2p, 128, 64);

    egcl_edge_kernel<<<(NE + 63) / 64, 256, 0, stream>>>(
        h, row, col, dist, eW1, eb1, eb2, eW1p, eW2p, agg);

    egcl_node_kernel<<<(NN + 63) / 64, 256, 0, stream>>>(
        h, agg, nb1, nb2, nW1p, nW2p, (float*)d_out);
}
